// EnergyConditionedAtomAttention_87479893885593
// MI455X (gfx1250) — matrix operand order repaired and checked by reading, not compiled, not run on hardware
//
#include <hip/hip_runtime.h>
#include <hip/hip_bf16.h>

// ---------------------------------------------------------------------------
// EnergyConditionedAtomAttention for MI455X (gfx1250, wave32, WMMA)
// B=8 N=64 NE=256 ATOM=128 E=64 RBF=32 HID=256 LATENT=256 HEADS=8 HD=32
// ---------------------------------------------------------------------------

#define BB 8
#define NN 64
#define NEE 256
#define ATOM_DIM 128
#define E_DIM 64
#define RBF_DIM 32
#define ZEMB 32
#define ABASE 196           // 128 + 32 + 32 + 3 + 1
#define CUT 6.0f
#define PI_F 3.14159265358979323846f

typedef __attribute__((ext_vector_type(16))) _Float16 v16h;
typedef __attribute__((ext_vector_type(8)))  _Float16 v8h;
typedef __attribute__((ext_vector_type(8)))  float    v8f;

#define LDK 264             // 256 + 8 halves padding (bank-conflict avoidance)

// fast silu: x * v_rcp_f32(1 + exp(-x))  -- avoids IEEE divide fixup chain
__device__ __forceinline__ float silu_f(float x) {
    return x * __builtin_amdgcn_rcpf(1.0f + __expf(-x));
}

// ---------------------------------------------------------------------------
// f32 -> f16 weight conversion
// ---------------------------------------------------------------------------
__global__ void cvt_f32_to_h_kernel(const float* __restrict__ s,
                                    _Float16* __restrict__ d, int n) {
    int i = blockIdx.x * blockDim.x + threadIdx.x;
    if (i < n) d[i] = (_Float16)s[i];
}

// ---------------------------------------------------------------------------
// Per-(b,n) featurization: atom_base, radial bias, valid, h_abs gather
// ---------------------------------------------------------------------------
__global__ void prep_kernel(const float* __restrict__ h,
                            const int* __restrict__ z,
                            const float* __restrict__ pos,
                            const unsigned char* __restrict__ mask,
                            const float* __restrict__ zemb,
                            const int* __restrict__ absI,
                            float* __restrict__ habs,
                            float* __restrict__ atom_base,
                            float* __restrict__ rbias,
                            float* __restrict__ validf) {
    int idx = blockIdx.x * blockDim.x + threadIdx.x;
    if (idx >= BB * NN) return;
    int b = idx / NN, n = idx % NN;
    int ai = absI[0];
    const float* p0 = pos + (b * NN + n) * 3;
    const float* pa = pos + (b * NN + ai) * 3;
    float dx = p0[0] - pa[0], dy = p0[1] - pa[1], dz = p0[2] - pa[2];
    float r = sqrtf(dx * dx + dy * dy + dz * dz);
    float ri = 1.0f / fmaxf(r, 1e-8f);
    float ux = dx * ri, uy = dy * ri, uz = dz * ri;
    int inCut = (r <= CUT) ? 1 : 0;
    int vld = (mask[idx] != 0) && inCut;

    float* ab = atom_base + (long)idx * ABASE;
    const float* hr = h + (long)idx * ATOM_DIM;
    for (int i = 0; i < ATOM_DIM; ++i) ab[i] = hr[i];
    const float* ze = zemb + z[idx] * ZEMB;
    for (int i = 0; i < ZEMB; ++i) ab[ATOM_DIM + i] = ze[i];
    const float delta = CUT / (float)(RBF_DIM - 1);
    const float gamma = 1.0f / (delta * delta + 1e-12f);
    float rc = fminf(r, CUT);
    for (int j = 0; j < RBF_DIM; ++j) {
        float d = rc - delta * (float)j;
        ab[ATOM_DIM + ZEMB + j] = __expf(-gamma * d * d);
    }
    ab[192] = ux; ab[193] = uy; ab[194] = uz;
    ab[195] = (n == ai) ? 1.0f : 0.0f;

    float cw = 0.5f * (cosf(PI_F * r / CUT) + 1.0f) * (inCut ? 1.0f : 0.0f);
    rbias[idx]  = logf(fmaxf(cw, 1e-8f));
    validf[idx] = vld ? 1.0f : 0.0f;

    if (n == 0) {  // gather h_abs for this batch
        const float* src = h + (long)(b * NN + ai) * ATOM_DIM;
        float* dst = habs + b * ATOM_DIM;
        for (int i = 0; i < ATOM_DIM; ++i) dst[i] = src[i];
    }
}

// ---------------------------------------------------------------------------
// Small exact f32 GEMM: out[r,c] = sum_k X[r,k] * W[wRowOff+k, c] (+bias[c])
// W is row-major with 256 columns. One thread per output element.
// ---------------------------------------------------------------------------
__global__ void dense_f32_kernel(const float* __restrict__ X, int ldx,
                                 const float* __restrict__ W, int wRowOff,
                                 const float* __restrict__ bias,
                                 float* __restrict__ out, int rows, int K) {
    int idx = blockIdx.x * blockDim.x + threadIdx.x;
    if (idx >= rows * 256) return;
    int r = idx >> 8, c = idx & 255;
    const float* x = X + (long)r * ldx;
    float acc = bias ? bias[c] : 0.0f;
    for (int k = 0; k < K; ++k) acc += x[k] * W[(long)(wRowOff + k) * 256 + c];
    out[idx] = acc;
}

// ---------------------------------------------------------------------------
// WMMA helpers (layouts per CDNA5 ISA 7.12.2, wave32)
// A tile 16x32 f16: lane L<16 -> row m0+L, K = {k0..k0+7, k0+16..k0+23}
//                   lane L>=16 -> row m0+L-16, K = {k0+8..k0+15, k0+24..k0+31}
// ---------------------------------------------------------------------------
__device__ __forceinline__ v16h load_a_lds(const _Float16* Xs, int m0, int k0,
                                           int lane) {
    int row = m0 + (lane & 15);
    int kb  = k0 + ((lane & 16) >> 1);          // +8 for upper lane half
    const _Float16* p = Xs + row * LDK + kb;
    v8h lo = *(const v8h*)(p);
    v8h hi = *(const v8h*)(p + 16);
    v16h a;
#pragma unroll
    for (int i = 0; i < 8; ++i) { a[i] = lo[i]; a[i + 8] = hi[i]; }
    return a;
}

// B tile 32x16 f16: lane L -> col n0+(L&15), K = {kb..kb+7, kb+16..kb+23}, kb = k0 + 8*(L>>4)
__device__ __forceinline__ v16h load_b_w(const _Float16* W, int k0, int n0,
                                         int lane) {
    int col = n0 + (lane & 15);
    int kb  = k0 + ((lane & 16) >> 1);
    v16h b;
#pragma unroll
    for (int i = 0; i < 8; ++i) {
        b[i]     = W[(long)(kb + i) * 256 + col];
        b[i + 8] = W[(long)(kb + 16 + i) * 256 + col];
    }
    return b;
}

// ---------------------------------------------------------------------------
// Generic fused 2-layer WMMA MLP over 128-row tiles (templated, branch-free).
//   X1[row,c] = f(preA[(g>>shiftA)*256+c] (+ preE[(g&255)*256+c]))  (f=silu opt)
//   Y  = silu(X1 @ W1 + b1)
//   O  = Y @ W2 + b2                -> f16 (OUT_HALF) or f32
// 256 threads = 8 waves; wave w owns N columns [32w, 32w+32).
// Dynamic LDS: 2 * 128 * LDK halves (~132 KB -> 2 blocks / WGP).
// ---------------------------------------------------------------------------
template <bool HAS_E, bool IN_SILU, bool OUT_HALF>
__global__ void mlp2_wmma_kernel(const float* __restrict__ preA, int shiftA,
                                 const float* __restrict__ preE,
                                 const _Float16* __restrict__ W1,
                                 const float* __restrict__ b1,
                                 const _Float16* __restrict__ W2,
                                 const float* __restrict__ b2,
                                 _Float16* __restrict__ outH,
                                 float* __restrict__ outF) {
    extern __shared__ _Float16 smem[];
    _Float16* Xs = smem;                 // 128 x LDK
    _Float16* Ys = smem + 128 * LDK;     // 128 x LDK

    const int tid = threadIdx.x;
    const int rowBase = blockIdx.x * 128;

    // --- build layer-1 activations in LDS (f16) ---
    for (int i = tid; i < 128 * 256; i += 256) {
        int r = i >> 8, c = i & 255;
        int g = rowBase + r;
        float v = preA[(long)(g >> shiftA) * 256 + c];
        if (HAS_E) v += preE[(long)(g & 255) * 256 + c];
        if (IN_SILU) v = silu_f(v);
        Xs[r * LDK + c] = (_Float16)v;
    }
    __syncthreads();

    const int wave = tid >> 5;
    const int lane = tid & 31;
    const int colLane = lane & 15;
    const int hi8 = (lane & 16) >> 1;   // +8 rows for upper half-wave

    // ---------------- layer 1: Y = silu(X @ W1 + b1) ----------------
    for (int nt = 0; nt < 2; ++nt) {
        int n0 = (wave * 2 + nt) * 16;
        v16h bregs[8];
#pragma unroll
        for (int ks = 0; ks < 8; ++ks) bregs[ks] = load_b_w(W1, ks * 32, n0, lane);
        float bias = b1[n0 + colLane];
        for (int mt = 0; mt < 8; ++mt) {
            int m0 = mt * 16;
            v8f acc = {};
#pragma unroll
            for (int ks = 0; ks < 8; ++ks) {
                v16h a = load_a_lds(Xs, m0, ks * 32, lane);
                acc = __builtin_amdgcn_wmma_f32_16x16x32_f16(
                    false, a, false, bregs[ks], (short)0, acc, false, false);
            }
#pragma unroll
            for (int rr = 0; rr < 8; ++rr) {
                float y = silu_f(acc[rr] + bias);
                Ys[(m0 + hi8 + rr) * LDK + n0 + colLane] = (_Float16)y;
            }
        }
    }
    __syncthreads();

    // ---------------- layer 2: O = Y @ W2 + b2 ----------------
    for (int nt = 0; nt < 2; ++nt) {
        int n0 = (wave * 2 + nt) * 16;
        v16h bregs[8];
#pragma unroll
        for (int ks = 0; ks < 8; ++ks) bregs[ks] = load_b_w(W2, ks * 32, n0, lane);
        float bias = b2[n0 + colLane];
        for (int mt = 0; mt < 8; ++mt) {
            int m0 = mt * 16;
            v8f acc = {};
#pragma unroll
            for (int ks = 0; ks < 8; ++ks) {
                v16h a = load_a_lds(Ys, m0, ks * 32, lane);
                acc = __builtin_amdgcn_wmma_f32_16x16x32_f16(
                    false, a, false, bregs[ks], (short)0, acc, false, false);
            }
            int col = n0 + colLane;
#pragma unroll
            for (int rr = 0; rr < 8; ++rr) {
                float y = acc[rr] + bias;
                long g = (long)(rowBase + m0 + hi8 + rr) * 256 + col;
                if (OUT_HALF) outH[g] = (_Float16)y;
                else          outF[g] = y;
            }
        }
    }
}

// ---------------------------------------------------------------------------
// Attention: one wave per (b,e). 8 heads x 32 dims; lane covers 8 contiguous
// dims => head = lane>>2; 4-lane shuffle reductions for dots & softmax.
// ---------------------------------------------------------------------------
__global__ void attention_kernel(const _Float16* __restrict__ qh,
                                 const _Float16* __restrict__ kh,
                                 const _Float16* __restrict__ vh,
                                 const float* __restrict__ rbias,
                                 const float* __restrict__ validf,
                                 float* __restrict__ attn_out) {
    __shared__ float sc[8][NN][8];     // [wave][n][head], 16 KB
    const int tid = threadIdx.x;
    const int wave = tid >> 5, lane = tid & 31;
    const int be = blockIdx.x * 8 + wave;   // 0..2047
    const int b = be >> 8, e = be & 255;
    const int hd = lane >> 2;
    const float scale = 0.17677669529663687f;   // 1/sqrt(32)

    float ql[8];
    {
        const _Float16* qp = qh + (long)be * 256 + lane * 8;
#pragma unroll
        for (int j = 0; j < 8; ++j) ql[j] = (float)qp[j];
    }

    // scores(n, head)
    for (int n = 0; n < NN; ++n) {
        const _Float16* kp = kh + ((long)((b * NN + n) * 256 + e)) * 256 + lane * 8;
        float p = 0.0f;
#pragma unroll
        for (int j = 0; j < 8; ++j) p += ql[j] * (float)kp[j];
        p += __shfl_xor(p, 1);
        p += __shfl_xor(p, 2);
        if ((lane & 3) == 0) {
            float vm = validf[b * NN + n];
            sc[wave][n][hd] = (vm != 0.0f) ? (p * scale + rbias[b * NN + n])
                                           : -1.0e9f;
        }
    }
    __syncthreads();

    // softmax over n per head (4 lanes x 16 n each)
    const int sub = lane & 3;
    float mx = -1.0e30f;
    for (int i = 0; i < 16; ++i) mx = fmaxf(mx, sc[wave][sub * 16 + i][hd]);
    mx = fmaxf(mx, __shfl_xor(mx, 1));
    mx = fmaxf(mx, __shfl_xor(mx, 2));
    float sum = 0.0f;
    for (int i = 0; i < 16; ++i) {
        float ex = __expf(sc[wave][sub * 16 + i][hd] - mx);
        sc[wave][sub * 16 + i][hd] = ex;
        sum += ex;
    }
    sum += __shfl_xor(sum, 1);
    sum += __shfl_xor(sum, 2);
    float inv = __builtin_amdgcn_rcpf(sum);
    __syncthreads();

    // out = sum_n attn(n,h) * v(n, :)
    float acc[8];
#pragma unroll
    for (int j = 0; j < 8; ++j) acc[j] = 0.0f;
    for (int n = 0; n < NN; ++n) {
        float a = sc[wave][n][hd] * inv * validf[b * NN + n];
        const _Float16* vp = vh + ((long)((b * NN + n) * 256 + e)) * 256 + lane * 8;
#pragma unroll
        for (int j = 0; j < 8; ++j) acc[j] += a * (float)vp[j];
    }
    float* op = attn_out + (long)be * 256 + lane * 8;
#pragma unroll
    for (int j = 0; j < 8; ++j) op[j] = acc[j];
}

// ---------------------------------------------------------------------------
// Host launcher
// ---------------------------------------------------------------------------
extern "C" void kernel_launch(void* const* d_in, const int* in_sizes, int n_in,
                              void* d_out, int out_size, void* d_ws, size_t ws_size,
                              hipStream_t stream) {
    (void)in_sizes; (void)n_in; (void)out_size; (void)ws_size;

    const float* h     = (const float*)d_in[0];
    const int*   z     = (const int*)d_in[1];
    const float* pos   = (const float*)d_in[2];
    const unsigned char* mask = (const unsigned char*)d_in[3];
    const float* efeat = (const float*)d_in[4];
    const int*   absI  = (const int*)d_in[5];
    const float* zemb  = (const float*)d_in[6];
    const float* qW0 = (const float*)d_in[7],  *qb0 = (const float*)d_in[8];
    const float* qW1 = (const float*)d_in[9],  *qb1 = (const float*)d_in[10];
    const float* qW2 = (const float*)d_in[11], *qb2 = (const float*)d_in[12];
    const float* kW0 = (const float*)d_in[13], *kb0 = (const float*)d_in[14];
    const float* kW1 = (const float*)d_in[15], *kb1 = (const float*)d_in[16];
    const float* kW2 = (const float*)d_in[17], *kb2 = (const float*)d_in[18];
    const float* vW0 = (const float*)d_in[19], *vb0 = (const float*)d_in[20];
    const float* vW1 = (const float*)d_in[21], *vb1 = (const float*)d_in[22];
    const float* vW2 = (const float*)d_in[23], *vb2 = (const float*)d_in[24];
    const float* oW0 = (const float*)d_in[25], *ob0 = (const float*)d_in[26];
    const float* oW1 = (const float*)d_in[27], *ob1 = (const float*)d_in[28];

    // ---- workspace layout ----
    char* ws = (char*)d_ws;
    size_t off = 0;
    auto alloc = [&](size_t bytes) {
        void* p = ws + off;
        off = (off + bytes + 255) & ~(size_t)255;
        return p;
    };
    float* habs      = (float*)alloc(BB * ATOM_DIM * 4);
    float* atom_base = (float*)alloc((size_t)BB * NN * ABASE * 4);
    float* rbias     = (float*)alloc(BB * NN * 4);
    float* validf    = (float*)alloc(BB * NN * 4);
    float* A1k = (float*)alloc((size_t)BB * NN * 256 * 4);
    float* A1v = (float*)alloc((size_t)BB * NN * 256 * 4);
    float* E1k = (float*)alloc((size_t)NEE * 256 * 4);
    float* E1v = (float*)alloc((size_t)NEE * 256 * 4);
    float* A1q = (float*)alloc((size_t)BB * 256 * 4);
    float* E1q = (float*)alloc((size_t)NEE * 256 * 4);
    _Float16* qW1h = (_Float16*)alloc(65536 * 2);
    _Float16* qW2h = (_Float16*)alloc(65536 * 2);
    _Float16* kW1h = (_Float16*)alloc(65536 * 2);
    _Float16* kW2h = (_Float16*)alloc(65536 * 2);
    _Float16* vW1h = (_Float16*)alloc(65536 * 2);
    _Float16* vW2h = (_Float16*)alloc(65536 * 2);
    _Float16* oW0h = (_Float16*)alloc(65536 * 2);
    _Float16* oW1h = (_Float16*)alloc(65536 * 2);
    _Float16* q_h  = (_Float16*)alloc((size_t)BB * NEE * 256 * 2);
    _Float16* k_h  = (_Float16*)alloc((size_t)BB * NN * NEE * 256 * 2);
    _Float16* v_h  = (_Float16*)alloc((size_t)BB * NN * NEE * 256 * 2);
    float* attn_out = (float*)alloc((size_t)BB * NEE * 256 * 4);

    // ---- 1) weight conversion to f16 ----
    {
        dim3 g(65536 / 256), blk(256);
        cvt_f32_to_h_kernel<<<g, blk, 0, stream>>>(qW1, qW1h, 65536);
        cvt_f32_to_h_kernel<<<g, blk, 0, stream>>>(qW2, qW2h, 65536);
        cvt_f32_to_h_kernel<<<g, blk, 0, stream>>>(kW1, kW1h, 65536);
        cvt_f32_to_h_kernel<<<g, blk, 0, stream>>>(kW2, kW2h, 65536);
        cvt_f32_to_h_kernel<<<g, blk, 0, stream>>>(vW1, vW1h, 65536);
        cvt_f32_to_h_kernel<<<g, blk, 0, stream>>>(vW2, vW2h, 65536);
        cvt_f32_to_h_kernel<<<g, blk, 0, stream>>>(oW0, oW0h, 65536);
        cvt_f32_to_h_kernel<<<g, blk, 0, stream>>>(oW1, oW1h, 65536);
    }

    // ---- 2) featurization ----
    prep_kernel<<<dim3(2), dim3(256), 0, stream>>>(h, z, pos, mask, zemb, absI,
                                                   habs, atom_base, rbias, validf);

    // ---- 3) layer-1 split GEMMs (exact f32) ----
    dense_f32_kernel<<<dim3(8),   dim3(256), 0, stream>>>(habs, ATOM_DIM, qW0, 0,   qb0, A1q, BB, ATOM_DIM);
    dense_f32_kernel<<<dim3(256), dim3(256), 0, stream>>>(efeat, E_DIM,  qW0, 128, nullptr, E1q, NEE, E_DIM);
    dense_f32_kernel<<<dim3(512), dim3(256), 0, stream>>>(atom_base, ABASE, kW0, 0, kb0, A1k, BB * NN, ABASE);
    dense_f32_kernel<<<dim3(256), dim3(256), 0, stream>>>(efeat, E_DIM,  kW0, ABASE, nullptr, E1k, NEE, E_DIM);
    dense_f32_kernel<<<dim3(512), dim3(256), 0, stream>>>(atom_base, ABASE, vW0, 0, vb0, A1v, BB * NN, ABASE);
    dense_f32_kernel<<<dim3(256), dim3(256), 0, stream>>>(efeat, E_DIM,  vW0, ABASE, nullptr, E1v, NEE, E_DIM);

    // ---- 4) WMMA MLPs ----
    const size_t ldsBytes = (size_t)2 * 128 * LDK * sizeof(_Float16);  // ~132 KB
    // Q: rows = B*NE = 2048
    mlp2_wmma_kernel<true, true, true><<<dim3(16), dim3(256), ldsBytes, stream>>>(
        A1q, 8, E1q, qW1h, qb1, qW2h, qb2, q_h, nullptr);
    // K: rows = B*N*NE = 131072
    mlp2_wmma_kernel<true, true, true><<<dim3(1024), dim3(256), ldsBytes, stream>>>(
        A1k, 8, E1k, kW1h, kb1, kW2h, kb2, k_h, nullptr);
    // V
    mlp2_wmma_kernel<true, true, true><<<dim3(1024), dim3(256), ldsBytes, stream>>>(
        A1v, 8, E1v, vW1h, vb1, vW2h, vb2, v_h, nullptr);

    // ---- 5) attention (softmax over atoms) ----
    attention_kernel<<<dim3(256), dim3(256), 0, stream>>>(
        q_h, k_h, v_h, rbias, validf, attn_out);

    // ---- 6) output MLP -> d_out (f32) ----
    mlp2_wmma_kernel<false, false, false><<<dim3(16), dim3(256), ldsBytes, stream>>>(
        attn_out, 0, nullptr, oW0h, ob0, oW1h, ob1, nullptr, (float*)d_out);
}
